// MoEFFN_31636729102523
// MI455X (gfx1250) — matrix operand order repaired and checked by reading, not compiled, not run on hardware
//
#include <hip/hip_runtime.h>
#include <math.h>

// MoE dense FFN, fused, bf16 WMMA + TDM (tensor_load_to_lds) double-buffered
// weight staging. gfx1250 / wave32.

typedef __attribute__((ext_vector_type(16))) __bf16 v16bf;
typedef __attribute__((ext_vector_type(8)))  float  v8f;
typedef __attribute__((ext_vector_type(4)))  unsigned int v4u;
typedef __attribute__((ext_vector_type(8)))  int  v8i;
typedef __attribute__((ext_vector_type(4)))  int  v4i;

#define HD   256
#define FD   512
#define NE   8
#define NTOK 32768
#define TM   64

// ---------------------------------------------------------------------------
// LDS byte offset of a __shared__ object (addrspace(3) ptrtoint).
// ---------------------------------------------------------------------------
__device__ inline unsigned lds_offset_of(const void* p)
{
    return (unsigned)(unsigned long long)
        (__attribute__((address_space(3))) const void*)p;
}

// ---------------------------------------------------------------------------
// Issue one 1-D TDM copy: ndwords dwords from gptr -> LDS byte offset.
// D# per CDNA5 ISA ch.8: group0 = {ctl, lds_addr, gaddr_lo, gaddr_hi|type},
// group1 carries data_size / tensor_dim0 / tile_dim0 / stride.
// Groups 2/3 disabled (<=2D tensor); trailing v8i operand zero-filled.
// ---------------------------------------------------------------------------
__device__ inline void tdm_load_1d(unsigned lds_byte_off, const void* gptr,
                                   unsigned ndwords)
{
    unsigned long long ga = (unsigned long long)gptr;
    v4u g0;
    g0[0] = 1u;                                        // count=1, user D#
    g0[1] = lds_byte_off;                              // lds_addr  [63:32]
    g0[2] = (unsigned)(ga & 0xFFFFFFFFu);              // gaddr     [95:64]
    g0[3] = (unsigned)((ga >> 32) & 0x1FFFFFFu)        // gaddr     [120:96]
          | (2u << 30);                                // type=2    [127:126]
    v8i g1;
    g1[0] = (int)(2u << 16);                           // data_size=4B, no mask
    g1[1] = (int)(ndwords << 16);                      // tensor_dim0[15:0] @48
    g1[2] = (int)((ndwords >> 16) | (1u << 16));       // dim0[31:16], dim1=1
    g1[3] = (int)(ndwords << 16);                      // tile_dim0 @[127:112]
    g1[4] = 0;                                         // tile_dim1/2 unused
    g1[5] = (int)ndwords;                              // dim0_stride[31:0]
    g1[6] = 0;
    g1[7] = 0;
    v4i gz4 = {0, 0, 0, 0};
    v8i gz8 = {0, 0, 0, 0, 0, 0, 0, 0};
    __builtin_amdgcn_tensor_load_to_lds(g0, g1, gz4, gz4, gz8, 0);
}

// ---------------------------------------------------------------------------
// Gate: logits = x @ Wg + bg ; softmax over E=8.  One thread per token.
// ---------------------------------------------------------------------------
__global__ __launch_bounds__(256) void moe_gates_kernel(
    const float* __restrict__ x, const float* __restrict__ Wg,
    const float* __restrict__ bg, float* __restrict__ gates)
{
    int n = blockIdx.x * 256 + threadIdx.x;
    if (n >= NTOK) return;
    float acc[NE];
#pragma unroll
    for (int e = 0; e < NE; ++e) acc[e] = bg[e];
    const float* xr = x + (size_t)n * HD;
    for (int h = 0; h < HD; ++h) {
        float xv = xr[h];
        const float* wr = Wg + h * NE;
#pragma unroll
        for (int e = 0; e < NE; ++e) acc[e] += xv * wr[e];
    }
    float mx = acc[0];
#pragma unroll
    for (int e = 1; e < NE; ++e) mx = fmaxf(mx, acc[e]);
    float sum = 0.f;
#pragma unroll
    for (int e = 0; e < NE; ++e) { acc[e] = __expf(acc[e] - mx); sum += acc[e]; }
    float inv = 1.0f / sum;
#pragma unroll
    for (int e = 0; e < NE; ++e) gates[n * NE + e] = acc[e] * inv;
}

// ---------------------------------------------------------------------------
// Pack W1 [E][H][F] f32 -> bf16 B-fragments: [e][nTileF=32][kTile=8][lane=32][16]
// B-fragment (32x16, 16-bit): lane = 16*half + n_local; element j -> k = (j<8) ? 8*half + j : 16 + 8*half + (j-8)
// ---------------------------------------------------------------------------
__global__ __launch_bounds__(256) void moe_pack_w1_kernel(
    const float* __restrict__ W1, __bf16* __restrict__ w1p)
{
    int idx  = blockIdx.x * 256 + threadIdx.x;      // 2^20 total
    int j    = idx & 15;
    int lane = (idx >> 4) & 31;
    int kt   = (idx >> 9) & 7;
    int nt   = (idx >> 12) & 31;
    int e    = idx >> 17;
    int half = lane >> 4;
    int k    = kt * 32 + ((j < 8) ? 8 * half + j : 16 + 8 * half + (j - 8)); // K within H
    int n    = nt * 16 + (lane & 15);               // column within F
    w1p[idx] = (__bf16)W1[((size_t)e * HD + k) * FD + n];
}

// ---------------------------------------------------------------------------
// Pack W2 [E][F][H] f32 -> bf16 B-fragments: [e][kTileF=16][nTileH=16][lane=32][16]
// ---------------------------------------------------------------------------
__global__ __launch_bounds__(256) void moe_pack_w2_kernel(
    const float* __restrict__ W2, __bf16* __restrict__ w2p)
{
    int idx  = blockIdx.x * 256 + threadIdx.x;      // 2^20 total
    int j    = idx & 15;
    int lane = (idx >> 4) & 31;
    int nt   = (idx >> 9) & 15;
    int kt   = (idx >> 13) & 15;
    int e    = idx >> 17;
    int half = lane >> 4;
    int k    = kt * 32 + ((j < 8) ? 8 * half + j : 16 + 8 * half + (j - 8)); // K within F
    int n    = nt * 16 + (lane & 15);               // column within H
    w2p[idx] = (__bf16)W2[((size_t)e * FD + k) * HD + n];
}

// ---------------------------------------------------------------------------
// Main fused kernel. 256 threads = 8 waves; 64 tokens per block.
// wave w: mTile = w>>1 (16 tokens), nHalf = w&1 (128 of 256 out columns).
// Weight chunks DMA'd by TDM with double buffering.
// ---------------------------------------------------------------------------
__global__ __launch_bounds__(256) void moe_main_kernel(
    const float* __restrict__ x, const float* __restrict__ gates,
    const __bf16* __restrict__ w1p, const __bf16* __restrict__ w2p,
    const float* __restrict__ b1, const float* __restrict__ b2,
    float* __restrict__ out)
{
    __shared__ alignas(32) __bf16 xsp[4 * 8 * 32 * 16];   // x A-frags   32 KB
    __shared__ alignas(32) __bf16 w1s[2][8192];           // W1 B-frags 2x16 KB
    __shared__ alignas(32) __bf16 w2s[2][8192];           // W2 B-frags 2x16 KB
    __shared__ alignas(32) __bf16 hsp[4 * 32 * 16];       // h  A-frags   4 KB

    const int tid  = threadIdx.x;
    const int lane = tid & 31;
    const int tok0 = blockIdx.x * TM;
    // scalar (SGPR) wave id -> clean s_cbranch around TDM issue
    const int waveS = __builtin_amdgcn_readfirstlane(tid) >> 5;

    // ---- load x tile [64 x 256], convert to bf16, store in A-fragment layout
    // A-fragment (16x32, 16-bit): lane = 16*half + m_local;
    //   element j: k = (j<8) ? 8*half + j : 16 + 8*half + (j-8)
    for (int i = tid; i < TM * HD; i += 256) {
        int m = i >> 8, k = i & 255;
        float xv = x[(size_t)(tok0 + m) * HD + k];
        int mt = m >> 4, ml = m & 15, kt = k >> 5, kk = k & 31;
        int h, j;
        if (kk < 16) { h = kk >> 3; j = kk & 7; }
        else         { h = (kk - 16) >> 3; j = 8 + ((kk - 16) & 7); }
        xsp[(((mt * 8 + kt) * 32) + (16 * h + ml)) * 16 + j] = (__bf16)xv;
    }

    // ---- kick off TDM for chunk 0 into buffer 0 (wave 0 only)
    if (waveS == 0) {
        tdm_load_1d(lds_offset_of(&w1s[0][0]), (const char*)w1p, 4096);
        tdm_load_1d(lds_offset_of(&w2s[0][0]), (const char*)w2p, 4096);
    }
    __syncthreads();

    const int wave    = tid >> 5;
    const int mTile   = wave >> 1;
    const int nHalf   = wave & 1;
    const int col     = lane & 15;
    const int rowHalf = lane >> 4;   // C/D tile: row M = r + 8*rowHalf

    // ---- hoist x A-fragments into registers (loop-invariant)
    v16bf xA[8];
#pragma unroll
    for (int kt = 0; kt < 8; ++kt)
        xA[kt] = *(const v16bf*)&xsp[((mTile * 8 + kt) * 32 + lane) * 16];

    v8f outAcc[8];
#pragma unroll
    for (int t = 0; t < 8; ++t) { v8f z = {}; outAcc[t] = z; }

    for (int e = 0; e < NE; ++e) {
        v8f yAcc[8];
#pragma unroll
        for (int t = 0; t < 8; ++t) {
            float b2v = b2[e * HD + nHalf * 128 + t * 16 + col];
            v8f tmp;
#pragma unroll
            for (int r = 0; r < 8; ++r) tmp[r] = b2v;
            yAcc[t] = tmp;
        }

        for (int fc = 0; fc < 16; ++fc) {
            const int c   = e * 16 + fc;       // flattened chunk id, 0..127
            const int buf = c & 1;

            // barrier A: everyone done reading buf[(c+1)&1] from chunk c-1
            __syncthreads();
            if (waveS == 0) {
                if (c + 1 < 128) {
                    // prefetch next chunk into the other buffer
                    tdm_load_1d(lds_offset_of(&w1s[buf ^ 1][0]),
                                (const char*)w1p + (size_t)(c + 1) * 16384, 4096);
                    tdm_load_1d(lds_offset_of(&w2s[buf ^ 1][0]),
                                (const char*)w2p + (size_t)(c + 1) * 16384, 4096);
                    __builtin_amdgcn_s_wait_tensorcnt((short)2); // chunk c landed
                } else {
                    __builtin_amdgcn_s_wait_tensorcnt((short)0);
                }
            }
            // barrier B: chunk c data visible to all waves
            __syncthreads();

            // ---- GEMM1: h tile (mTile, nHalf) = x[64,256] @ W1[:, 32-chunk]
            float b1v = b1[e * FD + fc * 32 + nHalf * 16 + col];
            v8f hAcc;
#pragma unroll
            for (int r = 0; r < 8; ++r) hAcc[r] = b1v;
#pragma unroll
            for (int kt = 0; kt < 8; ++kt) {
                v16bf b = *(const v16bf*)&w1s[buf][((nHalf * 8 + kt) * 32 + lane) * 16];
                hAcc = __builtin_amdgcn_wmma_f32_16x16x32_bf16(
                    false, xA[kt], false, b, (short)0, hAcc, false, false);
            }

            // ---- erf-GELU, convert to bf16, repack into h A-fragments
#pragma unroll
            for (int r = 0; r < 8; ++r) {
                float v = hAcc[r];
                float g = 0.5f * v * (1.0f + erff(v * 0.70710678118654752f));
                int M  = r + 8 * rowHalf;        // row within 16-token tile
                int kk = nHalf * 16 + col;       // column within 32-chunk
                int h, j;
                if (kk < 16) { h = kk >> 3; j = kk & 7; }
                else         { h = (kk - 16) >> 3; j = 8 + ((kk - 16) & 7); }
                hsp[((mTile * 32) + (16 * h + M)) * 16 + j] = (__bf16)g;
            }
            __syncthreads();   // barrier C: hsp ready

            // ---- GEMM2: y[16,128] += h_chunk[16,32] @ W2[32-chunk, 128]
            v16bf a2 = *(const v16bf*)&hsp[(mTile * 32 + lane) * 16];
#pragma unroll
            for (int t = 0; t < 8; ++t) {
                int nT2 = nHalf * 8 + t;
                v16bf b = *(const v16bf*)&w2s[buf][(nT2 * 32 + lane) * 16];
                yAcc[t] = __builtin_amdgcn_wmma_f32_16x16x32_bf16(
                    false, a2, false, b, (short)0, yAcc[t], false, false);
            }
        }

        // ---- out += gate[token][e] * y
        int baseTok = tok0 + mTile * 16 + 8 * rowHalf;
#pragma unroll
        for (int r = 0; r < 8; ++r) {
            float g = gates[(size_t)(baseTok + r) * NE + e];
#pragma unroll
            for (int t = 0; t < 8; ++t) outAcc[t][r] += g * yAcc[t][r];
        }
    }

    // ---- store result
    int baseTok = tok0 + mTile * 16 + 8 * rowHalf;
#pragma unroll
    for (int t = 0; t < 8; ++t) {
#pragma unroll
        for (int r = 0; r < 8; ++r) {
            out[(size_t)(baseTok + r) * HD + nHalf * 128 + t * 16 + col] = outAcc[t][r];
        }
    }
}

// ---------------------------------------------------------------------------
extern "C" void kernel_launch(void* const* d_in, const int* in_sizes, int n_in,
                              void* d_out, int out_size, void* d_ws, size_t ws_size,
                              hipStream_t stream)
{
    const float* x  = (const float*)d_in[0];
    const float* Wg = (const float*)d_in[1];
    const float* bg = (const float*)d_in[2];
    const float* W1 = (const float*)d_in[3];
    const float* b1 = (const float*)d_in[4];
    const float* W2 = (const float*)d_in[5];
    const float* b2 = (const float*)d_in[6];
    float* out = (float*)d_out;

    char* ws = (char*)d_ws;
    float*  gates = (float*)ws;                        // 32768*8*4 = 1 MB
    __bf16* w1p   = (__bf16*)(ws + (1u << 20));        // 2 MB packed bf16
    __bf16* w2p   = (__bf16*)(ws + 3u * (1u << 20));   // 2 MB packed bf16

    moe_gates_kernel<<<NTOK / 256, 256, 0, stream>>>(x, Wg, bg, gates);
    moe_pack_w1_kernel<<<(1u << 20) / 256, 256, 0, stream>>>(W1, w1p);
    moe_pack_w2_kernel<<<(1u << 20) / 256, 256, 0, stream>>>(W2, w2p);
    moe_main_kernel<<<NTOK / TM, 256, 0, stream>>>(x, gates, w1p, w2p, b1, b2, out);
}
